// Decoder_36825049596036
// MI455X (gfx1250) — matrix operand order repaired and checked by reading, not compiled, not run on hardware
//
#include <hip/hip_runtime.h>

typedef __attribute__((ext_vector_type(16))) __bf16 v16bf;
typedef __attribute__((ext_vector_type(8)))  float  v8f;

union V16 { v16bf bf; uint4 q[2]; };

__device__ __forceinline__ unsigned short f2bf1(float f) {
    unsigned int u = __float_as_uint(f);
    u += 0x7fffu + ((u >> 16) & 1u);      // round-to-nearest-even
    return (unsigned short)(u >> 16);
}

__device__ __forceinline__ unsigned int bfpk(float lo, float hi) {
    unsigned int ul = __float_as_uint(lo);
    unsigned int uh = __float_as_uint(hi);
    ul += 0x7fffu + ((ul >> 16) & 1u);
    uh += 0x7fffu + ((uh >> 16) & 1u);
    return (ul >> 16) | (uh & 0xffff0000u);
}

// ---------------------------------------------------------------------------
// one-time packs (conversion lives here, not in the GEMM hot loops)
// ---------------------------------------------------------------------------

// fp32 -> bf16, row-major passthrough (A-side). Exact grids (total/4/256).
__global__ __launch_bounds__(256)
void pack_a(const float* __restrict__ src, unsigned short* __restrict__ dst)
{
    const int i = blockIdx.x * 256 + threadIdx.x;
    float4 v = ((const float4*)src)[i];
    uint2 p;
    p.x = bfpk(v.x, v.y);
    p.y = bfpk(v.z, v.w);
    ((uint2*)dst)[i] = p;
}

// fp32 [K][N] row-major -> packed KN32 bf16:
// element (k,n) at ((k/32)*N + n)*32 + (k%32). Transpose via LDS.
__global__ __launch_bounds__(256)
void pack_b_kn32(const float* __restrict__ Bm, unsigned short* __restrict__ out, int N)
{
    __shared__ __align__(16) unsigned short T[128 * 32];
    const int kb  = blockIdx.y;
    const int n0  = blockIdx.x * 128;
    const int tid = threadIdx.x;
    #pragma unroll
    for (int i = 0; i < 2; ++i) {
        const int p  = tid + i * 256;
        const int kk = (p >> 5) * 2, nn = (p & 31) * 4;
        const float* src = Bm + (size_t)(kb * 32 + kk) * N + n0 + nn;
        float4 r0 = *(const float4*)src;
        float4 r1 = *(const float4*)(src + N);
        *(unsigned int*)&T[(nn + 0) * 32 + kk] = bfpk(r0.x, r1.x);
        *(unsigned int*)&T[(nn + 1) * 32 + kk] = bfpk(r0.y, r1.y);
        *(unsigned int*)&T[(nn + 2) * 32 + kk] = bfpk(r0.z, r1.z);
        *(unsigned int*)&T[(nn + 3) * 32 + kk] = bfpk(r0.w, r1.w);
    }
    __syncthreads();
    unsigned short* dst = out + ((size_t)kb * N + n0) * 32;
    const int c = tid >> 1, part = tid & 1;
    const uint4* s = (const uint4*)&T[c * 32 + part * 16];
    uint4* d = (uint4*)(dst + c * 32 + part * 16);
    d[0] = s[0];
    d[1] = s[1];
}

// ---------------------------------------------------------------------------
// bf16-WMMA GEMM, operands pre-packed in global; no LDS staging, no barriers
// in the K loop. Workgroup tile 64(M) x 128(N), 8 wave32s (2Mx4N), each wave
// owns 2Mx2N = four 16x16 WMMA tiles. Software-pipelined: fragments for
// K-block kb+1 load into fresh registers while block kb's 4 WMMAs issue
// (8x global_load_b128 : 4x v_wmma per lane-step).
// MODE 0: C = A@B + bias0[n] (+bias1[n])
// MODE 1: logits[m] += sum_n tanh(acc + bias0[n] + hproj[b,n]) * Vvec[n]
// ---------------------------------------------------------------------------
template <int MODE>
__global__ __launch_bounds__(256)
void gemm_wmma(const unsigned short* __restrict__ Abf, int lda,
               const unsigned short* __restrict__ Bpk, int N,
               const float* __restrict__ bias0, const float* __restrict__ bias1,
               float* __restrict__ C, int ldc, int K32,
               const float* __restrict__ hproj, int ldh,
               const float* __restrict__ Vvec,
               float* __restrict__ logits, int seqS)
{
    __shared__ float sh_logit[64];
    const int tid = threadIdx.x;
    const int m0  = blockIdx.y * 64;
    const int n0  = blockIdx.x * 128;
    const int wv = tid >> 5, lane = tid & 31;
    const int mw = wv & 1, nwav = wv >> 1;
    const int lr = lane & 15, hi = lane >> 4;

    if (MODE == 1) {
        if (tid < 64) sh_logit[tid] = 0.f;
        __syncthreads();
    }

    const int m  = m0 + mw * 32 + lr;       // first of two 16-row subtiles
    const int nA = n0 + nwav * 16 + lr;
    const int nB = nA + 64;
    const unsigned short* pA0 = Abf + (size_t)m * lda + hi * 16;
    const unsigned short* pA1 = pA0 + (size_t)16 * lda;
    const unsigned short* pBA = Bpk + (size_t)nA * 32 + hi * 16;
    const unsigned short* pBB = Bpk + (size_t)nB * 32 + hi * 16;
    const size_t bstep = (size_t)N * 32;

    v8f acc00 = {}, acc01 = {}, acc10 = {}, acc11 = {};

    // preload K-block 0
    V16 a0, a1, b, c;
    a0.q[0] = *(const uint4*)pA0;  a0.q[1] = *(const uint4*)(pA0 + 8);  pA0 += 32;
    a1.q[0] = *(const uint4*)pA1;  a1.q[1] = *(const uint4*)(pA1 + 8);  pA1 += 32;
    b.q[0]  = *(const uint4*)pBA;  b.q[1]  = *(const uint4*)(pBA + 8);   pBA += bstep;
    c.q[0]  = *(const uint4*)pBB;  c.q[1]  = *(const uint4*)(pBB + 8);   pBB += bstep;

    for (int kb = 0; kb < K32 - 1; ++kb) {
        // issue next K-block loads into fresh registers
        V16 na0, na1, nb, nc;
        na0.q[0] = *(const uint4*)pA0;  na0.q[1] = *(const uint4*)(pA0 + 8);  pA0 += 32;
        na1.q[0] = *(const uint4*)pA1;  na1.q[1] = *(const uint4*)(pA1 + 8);  pA1 += 32;
        nb.q[0]  = *(const uint4*)pBA;  nb.q[1]  = *(const uint4*)(pBA + 8);   pBA += bstep;
        nc.q[0]  = *(const uint4*)pBB;  nc.q[1]  = *(const uint4*)(pBB + 8);   pBB += bstep;
        __builtin_prefetch(pBA + bstep, 0, 3);   // ~2 K-blocks ahead of the B stream
        __builtin_prefetch(pBB + bstep, 0, 3);

        // consume current K-block
        acc00 = __builtin_amdgcn_wmma_f32_16x16x32_bf16(
            false, a0.bf, false, b.bf, (short)0, acc00, false, false);
        acc01 = __builtin_amdgcn_wmma_f32_16x16x32_bf16(
            false, a0.bf, false, c.bf, (short)0, acc01, false, false);
        acc10 = __builtin_amdgcn_wmma_f32_16x16x32_bf16(
            false, a1.bf, false, b.bf, (short)0, acc10, false, false);
        acc11 = __builtin_amdgcn_wmma_f32_16x16x32_bf16(
            false, a1.bf, false, c.bf, (short)0, acc11, false, false);

        a0 = na0; a1 = na1; b = nb; c = nc;
    }
    // last K-block
    acc00 = __builtin_amdgcn_wmma_f32_16x16x32_bf16(
        false, a0.bf, false, b.bf, (short)0, acc00, false, false);
    acc01 = __builtin_amdgcn_wmma_f32_16x16x32_bf16(
        false, a0.bf, false, c.bf, (short)0, acc01, false, false);
    acc10 = __builtin_amdgcn_wmma_f32_16x16x32_bf16(
        false, a1.bf, false, b.bf, (short)0, acc10, false, false);
    acc11 = __builtin_amdgcn_wmma_f32_16x16x32_bf16(
        false, a1.bf, false, c.bf, (short)0, acc11, false, false);

    const int mbase = m0 + mw * 32 + hi * 8;
    if (MODE == 0) {
        const float bA = (bias0 ? bias0[nA] : 0.f) + (bias1 ? bias1[nA] : 0.f);
        const float bB = (bias0 ? bias0[nB] : 0.f) + (bias1 ? bias1[nB] : 0.f);
        float* Crow = C + (size_t)mbase * ldc;
        #pragma unroll
        for (int r = 0; r < 8; ++r) {
            Crow[nA] = acc00[r] + bA;
            Crow[nB] = acc01[r] + bB;
            Crow += ldc;
        }
        Crow = C + (size_t)(mbase + 16) * ldc;
        #pragma unroll
        for (int r = 0; r < 8; ++r) {
            Crow[nA] = acc10[r] + bA;
            Crow[nB] = acc11[r] + bB;
            Crow += ldc;
        }
    } else {
        // all 64 rows in this tile share one batch index (m0 % 64 == 0, S = 64)
        const int bidx = m0 / seqS;
        const float addA = bias0[nA] + hproj[(size_t)bidx * ldh + nA];
        const float addB = bias0[nB] + hproj[(size_t)bidx * ldh + nB];
        const float vA = Vvec[nA], vB = Vvec[nB];
        const int mloc = mw * 32 + hi * 8;
        #pragma unroll
        for (int r = 0; r < 8; ++r) {
            float v0 = tanhf(acc00[r] + addA) * vA + tanhf(acc01[r] + addB) * vB;
            float v1 = tanhf(acc10[r] + addA) * vA + tanhf(acc11[r] + addB) * vB;
            atomicAdd(&sh_logit[mloc + r], v0);
            atomicAdd(&sh_logit[mloc + 16 + r], v1);
        }
        __syncthreads();
        if (tid < 64) atomicAdd(&logits[m0 + tid], sh_logit[tid]);
    }
}

// ---------------------------------------------------------------------------
// embedding gather (bf16 into GRU input): inp_bf[b, 1024+e] = emb[x[b], e]
// ---------------------------------------------------------------------------
__global__ void embed_kernel(const int* __restrict__ x,
                             const float* __restrict__ emb,
                             unsigned short* __restrict__ inp_bf)
{
    const int b = blockIdx.x, e = threadIdx.x;
    inp_bf[(size_t)b * 1280 + 1024 + e] = f2bf1(emb[(size_t)x[b] * 256 + e]);
}

// ---------------------------------------------------------------------------
// softmax over S=64 per batch row (bv cancels); ws copy + output copy
// ---------------------------------------------------------------------------
__global__ void softmax64(const float* __restrict__ logits,
                          float* __restrict__ attn,
                          float* __restrict__ out_attn)
{
    __shared__ float buf[64];
    const int b = blockIdx.x, t = threadIdx.x;
    const float v = logits[b * 64 + t];
    buf[t] = v; __syncthreads();
    for (int off = 32; off > 0; off >>= 1) {
        if (t < off) buf[t] = fmaxf(buf[t], buf[t + off]);
        __syncthreads();
    }
    const float mx = buf[0]; __syncthreads();
    const float e = expf(v - mx);
    buf[t] = e; __syncthreads();
    for (int off = 32; off > 0; off >>= 1) {
        if (t < off) buf[t] += buf[t + off];
        __syncthreads();
    }
    const float w = e / buf[0];
    attn[b * 64 + t] = w;
    out_attn[b * 64 + t] = w;
}

// ---------------------------------------------------------------------------
// context[b,u] = sum_s w[b,s] * enc[b,s,u] -> bf16 GRU input (cols 0..1023)
// ---------------------------------------------------------------------------
__global__ __launch_bounds__(256)
void context_kernel(const float* __restrict__ attn,
                    const float* __restrict__ enc,
                    unsigned short* __restrict__ inp_bf)
{
    __shared__ float w[64];
    const int b = blockIdx.y;
    const int u = blockIdx.x * 256 + threadIdx.x;
    if (threadIdx.x < 64) w[threadIdx.x] = attn[b * 64 + threadIdx.x];
    __syncthreads();
    const float* eb = enc + (size_t)b * 64 * 1024 + u;
    float acc = 0.f;
    #pragma unroll 8
    for (int s = 0; s < 64; ++s) acc += w[s] * eb[(size_t)s * 1024];
    inp_bf[(size_t)b * 1280 + u] = f2bf1(acc);
}

// ---------------------------------------------------------------------------
// GRU pointwise: h0 == 0 so hk = gru_bias[1]; state = (1-z)*hh
// xk already contains gru_bias[0]. Writes f32 state (output) + bf16 (FC A).
// ---------------------------------------------------------------------------
__global__ __launch_bounds__(256)
void gru_pointwise(const float* __restrict__ xk,
                   const float* __restrict__ gbias,
                   unsigned short* __restrict__ state_bf,
                   float* __restrict__ out_state)
{
    const int idx = blockIdx.x * 256 + threadIdx.x;   // B*U = 262144
    const int b = idx >> 10, u = idx & 1023;
    const float* gb1 = gbias + 3072;                  // gru_bias[1]
    const float xz = xk[(size_t)b * 3072 + u]        + gb1[u];
    const float xr = xk[(size_t)b * 3072 + 1024 + u] + gb1[1024 + u];
    const float xh = xk[(size_t)b * 3072 + 2048 + u];
    const float rh = gb1[2048 + u];
    const float z  = 1.f / (1.f + expf(-xz));
    const float r  = 1.f / (1.f + expf(-xr));
    const float hh = tanhf(xh + r * rh);
    const float st = (1.f - z) * hh;
    out_state[idx] = st;
    state_bf[idx]  = f2bf1(st);
}

// ---------------------------------------------------------------------------
extern "C" void kernel_launch(void* const* d_in, const int* in_sizes, int n_in,
                              void* d_out, int out_size, void* d_ws, size_t ws_size,
                              hipStream_t stream)
{
    (void)in_sizes; (void)n_in; (void)out_size; (void)ws_size;
    const int*   x      = (const int*)  d_in[0];
    const float* hidden = (const float*)d_in[1];
    const float* enc    = (const float*)d_in[2];
    const float* emb    = (const float*)d_in[3];
    const float* W1     = (const float*)d_in[4];
    const float* b1     = (const float*)d_in[5];
    const float* W2     = (const float*)d_in[6];
    const float* b2     = (const float*)d_in[7];
    const float* Vv     = (const float*)d_in[8];
    // d_in[9]  bv: uniform logit shift, cancels in softmax -> unused
    const float* gk     = (const float*)d_in[10];
    // d_in[11] gru_rec_kernel: h0 == 0 -> recurrent GEMM is pure bias -> unused
    const float* gb     = (const float*)d_in[12];
    const float* fcW    = (const float*)d_in[13];
    const float* fcb    = (const float*)d_in[14];

    // ---- workspace carve-out ----
    float* wsf    = (float*)d_ws;
    float* hproj  = wsf;                 // 262144
    float* logits = hproj  + 262144;     // 16384
    float* attn   = logits + 16384;      // 16384
    float* xk     = attn   + 16384;      // 786432
    unsigned short* h        = (unsigned short*)(xk + 786432);
    unsigned short* enc_bf   = h;  h += 16777216;   // 16384 x 1024
    unsigned short* hid_bf   = h;  h += 262144;     // 256 x 1024
    unsigned short* inp_bf   = h;  h += 327680;     // 256 x 1280
    unsigned short* state_bf = h;  h += 262144;     // 256 x 1024
    unsigned short* W1pk     = h;  h += 1048576;    // KN32 1024x1024
    unsigned short* W2pk     = h;  h += 1048576;    // KN32 1024x1024
    unsigned short* grupk    = h;  h += 3932160;    // KN32 1280x3072
    unsigned short* fcWpk    = h;                   // KN32 1024x32000

    float* out       = (float*)d_out;
    float* out_state = out + (size_t)256 * 32000;
    float* out_attn  = out_state + 256 * 1024;

    hipMemsetAsync(logits, 0, 16384 * sizeof(float), stream);

    // ---- one-time bf16 packs ----
    pack_a<<<16384, 256, 0, stream>>>(enc, enc_bf);       // 16.8M elems
    pack_a<<<256,   256, 0, stream>>>(hidden, hid_bf);    // 262144 elems
    pack_b_kn32<<<dim3(8,  32), 256, 0, stream>>>(W2,  W2pk,  1024);
    pack_b_kn32<<<dim3(8,  32), 256, 0, stream>>>(W1,  W1pk,  1024);
    pack_b_kn32<<<dim3(24, 40), 256, 0, stream>>>(gk,  grupk, 3072);
    pack_b_kn32<<<dim3(250,32), 256, 0, stream>>>(fcW, fcWpk, 32000);

    embed_kernel<<<256, 256, 0, stream>>>(x, emb, inp_bf);

    // hproj = hidden @ W2 + b2 + b1            (M=256, N=1024, K=1024)
    gemm_wmma<0><<<dim3(8, 4), 256, 0, stream>>>(
        hid_bf, 1024, W2pk, 1024, b2, b1, hproj, 1024, 32,
        nullptr, 0, nullptr, nullptr, 64);

    // fused score+logit: tanh(enc@W1 + b1 + hproj)·V  (M=16384, N=1024, K=1024)
    gemm_wmma<1><<<dim3(8, 256), 256, 0, stream>>>(
        enc_bf, 1024, W1pk, 1024, b1, nullptr, nullptr, 0, 32,
        hproj, 1024, Vv, logits, 64);

    softmax64<<<256, 64, 0, stream>>>(logits, attn, out_attn);

    context_kernel<<<dim3(4, 256), 256, 0, stream>>>(attn, enc, inp_bf);

    // xk = inp @ gru_kernel + gru_bias[0]      (M=256, N=3072, K=1280)
    gemm_wmma<0><<<dim3(24, 4), 256, 0, stream>>>(
        inp_bf, 1280, grupk, 3072, gb, nullptr, xk, 3072, 40,
        nullptr, 0, nullptr, nullptr, 64);

    gru_pointwise<<<1024, 256, 0, stream>>>(xk, gb, state_bf, out_state);

    // out = state @ fc_W + fc_b               (M=256, N=32000, K=1024)
    gemm_wmma<0><<<dim3(250, 4), 256, 0, stream>>>(
        state_bf, 1024, fcWpk, 32000, fcb, nullptr, out, 32000, 32,
        nullptr, 0, nullptr, nullptr, 64);
}
